// TransformerBlock_48473000902943
// MI455X (gfx1250) — matrix operand order repaired and checked by reading, not compiled, not run on hardware
//
#include <hip/hip_runtime.h>
#include <hip/hip_bf16.h>
#include <math.h>

// ---------------------------------------------------------------------------
// MI455X (gfx1250) fused transformer block.
// ~185 GFLOP vs ~hundreds of MB moved -> matrix-engine bound.
// bf16 operands + f32 accumulation via v_wmma_f32_16x16x32_bf16.
// GEMMs use 32-row M-tiles so each B fragment feeds two WMMAs (2:1
// compute:load), halving fragment bandwidth per FLOP.
// ---------------------------------------------------------------------------

typedef __attribute__((ext_vector_type(16))) __bf16 bf16x16;
typedef __attribute__((ext_vector_type(8)))  __bf16 bf16x8;
typedef __attribute__((ext_vector_type(8)))  float  f32x8;

#define B_  4
#define S_  4096
#define E_  512
#define F_  2048
#define M_  (B_ * S_)

#define WMMA_BF16(a_, b_, c_) \
  __builtin_amdgcn_wmma_f32_16x16x32_bf16(false, (a_), false, (b_), (short)0, (c_), false, false)

__device__ __forceinline__ __bf16 f2bf(float f) {
  unsigned u = __builtin_bit_cast(unsigned, f);
  unsigned r = u + 0x7fffu + ((u >> 16) & 1u);   // round-to-nearest-even
  unsigned short h = (unsigned short)(r >> 16);
  return __builtin_bit_cast(__bf16, h);
}

// A-operand fragment (16x32, M x K). Row-major source, ld in elements.
// lane<16: M=lane, K = {k0+0..7, k0+16..23}; lane>=16: M=lane-16, K={k0+8..15, k0+24..31}
__device__ __forceinline__ bf16x16 load_frag_a(const __bf16* base, int ld, int row0, int k0) {
  const int lane = threadIdx.x & 31;
  const int hlf  = lane >> 4;
  const __bf16* p = base + (size_t)(row0 + (lane & 15)) * ld + k0 + hlf * 8;
  bf16x16 f;
  *((bf16x8*)&f)       = *((const bf16x8*)p);
  *(((bf16x8*)&f) + 1) = *((const bf16x8*)(p + 16));
  return f;
}

// B-operand fragment (32x16, K x N) read from "N-major" storage (row n holds K
// contiguously), i.e. B[k][n] = src[n][k].  lane<16: N=lane, K={k0+0..7, k0+16..23};
// lane>=16: N=lane-16, K={k0+8..15, k0+24..31}.
__device__ __forceinline__ bf16x16 load_frag_b(const __bf16* base, int ld, int n0, int k0) {
  const int lane = threadIdx.x & 31;
  const int hlf  = lane >> 4;
  const __bf16* p = base + (size_t)(n0 + (lane & 15)) * ld + k0 + hlf * 8;
  bf16x16 f;
  *((bf16x8*)&f)       = *((const bf16x8*)p);
  *(((bf16x8*)&f) + 1) = *((const bf16x8*)(p + 16));
  return f;
}

// ---------------------------------------------------------------------------
// f32 -> bf16 conversion
// ---------------------------------------------------------------------------
__global__ void cvt_f32_bf16(const float* __restrict__ in, __bf16* __restrict__ out, int n) {
  int i = blockIdx.x * blockDim.x + threadIdx.x;
  if (i < n) out[i] = f2bf(in[i]);
}

// ---------------------------------------------------------------------------
// QKV projection: out[m][n] = sum_k x[m][k] * W[n][k]   (x @ W^T)
// grid (M/32, 512/256, 3), block 128.  32-row M-tile: each B fragment is
// consumed by two WMMAs, so next fragment's loads overlap the second WMMA.
// z selects q/k/v.  V stored transposed.
// ---------------------------------------------------------------------------
__global__ void qkv_gemm(const __bf16* __restrict__ xb,
                         const __bf16* __restrict__ Wqb,
                         const __bf16* __restrict__ Wkb,
                         const __bf16* __restrict__ Wvb,
                         __bf16* __restrict__ qb,
                         __bf16* __restrict__ kb,
                         __bf16* __restrict__ vT) {
  const int lane = threadIdx.x & 31;
  const int wave = threadIdx.x >> 5;
  const int hlf  = lane >> 4;
  const int lidx = lane & 15;
  const int m0 = blockIdx.x * 32;
  const int n0 = blockIdx.y * 256 + wave * 64;
  const int which = blockIdx.z;
  const __bf16* W = (which == 0) ? Wqb : (which == 1) ? Wkb : Wvb;

  f32x8 zero = {};
  f32x8 acc[2][4];
#pragma unroll
  for (int r = 0; r < 2; ++r)
#pragma unroll
    for (int t = 0; t < 4; ++t) acc[r][t] = zero;

#pragma unroll
  for (int k = 0; k < E_; k += 32) {
    bf16x16 a0 = load_frag_a(xb, E_, m0,      k);
    bf16x16 a1 = load_frag_a(xb, E_, m0 + 16, k);
#pragma unroll
    for (int t = 0; t < 4; ++t) {
      bf16x16 bb = load_frag_b(W, E_, n0 + t * 16, k);
      acc[0][t] = WMMA_BF16(a0, bb, acc[0][t]);
      acc[1][t] = WMMA_BF16(a1, bb, acc[1][t]);
    }
  }

  if (which < 2) {
    __bf16* dst = (which == 0) ? qb : kb;
#pragma unroll
    for (int r = 0; r < 2; ++r)
#pragma unroll
      for (int t = 0; t < 4; ++t) {
        int col = n0 + t * 16 + lidx;
#pragma unroll
        for (int i = 0; i < 8; ++i)
          dst[(size_t)(m0 + r * 16 + hlf * 8 + i) * E_ + col] = f2bf(acc[r][t][i]);
      }
  } else {
    // vT[b][e][s]
    const int b  = m0 >> 12;           // S_ = 4096
    const int s0 = m0 & (S_ - 1);
    __bf16* dst = vT + (size_t)b * E_ * S_;
#pragma unroll
    for (int r = 0; r < 2; ++r)
#pragma unroll
      for (int t = 0; t < 4; ++t) {
        int e = n0 + t * 16 + lidx;
#pragma unroll
        for (int i = 0; i < 8; ++i)
          dst[(size_t)e * S_ + (s0 + r * 16 + hlf * 8 + i)] = f2bf(acc[r][t][i]);
      }
  }
}

// ---------------------------------------------------------------------------
// Flash attention over one 16-row query tile per workgroup (128 threads /
// 4 waves).  Waves split the e-dimension (128 cols each).  Fused: post-softmax
// 1/sqrt(E) scale, residual +x, LayerNorm-1.  Writes mid (f32 + bf16).
// ---------------------------------------------------------------------------
__global__ void attn_flash_ln(const __bf16* __restrict__ qb,
                              const __bf16* __restrict__ kb,
                              const __bf16* __restrict__ vT,
                              const float* __restrict__ x,
                              const float* __restrict__ g1,
                              const float* __restrict__ bet1,
                              float* __restrict__ midf,
                              __bf16* __restrict__ midb) {
  const int lane = threadIdx.x & 31;
  const int wave = threadIdx.x >> 5;
  const int hlf  = lane >> 4;
  const int lidx = lane & 15;
  const int m0 = blockIdx.x * 16;      // global query row base
  const int b  = m0 >> 12;
  const int key_base = b * S_;
  const __bf16* vTb = vT + (size_t)b * E_ * S_;

  __shared__ __align__(16) float  spart[4][512];   // per-wave partial scores
  __shared__ __align__(16) float  sfin[512];       // reduced scores, col-major [col][row]
  __shared__ __align__(16) __bf16 p_s[16 * 32];    // softmax weights tile (A operand)
  __shared__ float alpha_s[16];
  __shared__ float lrow_s[16];
  __shared__ float rsum[16], rsq[16];

  // q fragments for this wave's e-slice [wave*128, wave*128+128)
  bf16x16 qa[4];
#pragma unroll
  for (int kk = 0; kk < 4; ++kk)
    qa[kk] = load_frag_a(qb, E_, m0, wave * 128 + kk * 32);

  f32x8 zero = {};
  f32x8 oacc[8];
#pragma unroll
  for (int t = 0; t < 8; ++t) oacc[t] = zero;

  float m_run = -3.0e38f, l_run = 0.0f;   // live only in wave0 lanes<16

  for (int kt = 0; kt < S_; kt += 32) {
    // ---- partial scores S = q_slice @ k_slice^T  (16 x 32) ----
    // Issue all fragment loads for a tile first, then the WMMA chain.
#pragma unroll
    for (int nt = 0; nt < 2; ++nt) {
      bf16x16 kf[4];
#pragma unroll
      for (int kk = 0; kk < 4; ++kk)
        kf[kk] = load_frag_b(kb, E_, key_base + kt + nt * 16, wave * 128 + kk * 32);
      f32x8 s = zero;
#pragma unroll
      for (int kk = 0; kk < 4; ++kk)
        s = WMMA_BF16(qa[kk], kf[kk], s);
#pragma unroll
      for (int i = 0; i < 8; ++i)
        spart[wave][(nt * 16 + lidx) * 16 + hlf * 8 + i] = s[i];
    }
    __syncthreads();

    // ---- reduce partials across waves ----
    {
      int tid = threadIdx.x;
#pragma unroll
      for (int j = 0; j < 4; ++j) {
        int idx = tid + j * 128;
        sfin[idx] = spart[0][idx] + spart[1][idx] + spart[2][idx] + spart[3][idx];
      }
    }
    __syncthreads();

    // ---- online softmax (wave0, one lane per query row) ----
    if (wave == 0 && lane < 16) {
      int r = lane;
      float sc[32];
      float mt = -3.0e38f;
#pragma unroll
      for (int c = 0; c < 32; ++c) { sc[c] = sfin[c * 16 + r]; mt = fmaxf(mt, sc[c]); }
      float m_new = fmaxf(m_run, mt);
      float alpha = __expf(m_run - m_new);
      float psum = 0.0f;
#pragma unroll
      for (int c = 0; c < 32; ++c) {
        float p = __expf(sc[c] - m_new);
        psum += p;
        p_s[r * 32 + c] = f2bf(p);
      }
      l_run = l_run * alpha + psum;
      m_run = m_new;
      alpha_s[r] = alpha;
    }
    __syncthreads();

    // ---- rescale accumulators, accumulate P @ V for this wave's e-cols ----
    float al[8];
#pragma unroll
    for (int i = 0; i < 8; ++i) al[i] = alpha_s[hlf * 8 + i];
    bf16x16 pa = load_frag_a(p_s, 32, 0, 0);
    // Pairs: V-fragment loads overlap with rescale VALU + previous WMMAs.
#pragma unroll
    for (int t = 0; t < 8; t += 2) {
      bf16x16 vf0 = load_frag_b(vTb, S_, wave * 128 + t * 16, kt);
      bf16x16 vf1 = load_frag_b(vTb, S_, wave * 128 + (t + 1) * 16, kt);
#pragma unroll
      for (int i = 0; i < 8; ++i) { oacc[t][i] *= al[i]; oacc[t + 1][i] *= al[i]; }
      oacc[t]     = WMMA_BF16(pa, vf0, oacc[t]);
      oacc[t + 1] = WMMA_BF16(pa, vf1, oacc[t + 1]);
    }
    __syncthreads();
  }

  // ---- finalize: /(l*sqrt(E)), residual, LayerNorm-1 ----
  if (wave == 0 && lane < 16)
    lrow_s[lane] = 1.0f / (l_run * 22.627416997969522f);   // sqrt(512)
  if (threadIdx.x < 16) { rsum[threadIdx.x] = 0.0f; rsq[threadIdx.x] = 0.0f; }
  __syncthreads();

  float li[8];
#pragma unroll
  for (int i = 0; i < 8; ++i) li[i] = lrow_s[hlf * 8 + i];
#pragma unroll
  for (int t = 0; t < 8; ++t) {
    int col = wave * 128 + t * 16 + lidx;
#pragma unroll
    for (int i = 0; i < 8; ++i) {
      int row = m0 + hlf * 8 + i;
      oacc[t][i] = oacc[t][i] * li[i] + x[(size_t)row * E_ + col];
    }
  }
#pragma unroll
  for (int i = 0; i < 8; ++i) {
    float s = 0.0f, q = 0.0f;
#pragma unroll
    for (int t = 0; t < 8; ++t) { float v = oacc[t][i]; s += v; q += v * v; }
    atomicAdd(&rsum[hlf * 8 + i], s);
    atomicAdd(&rsq[hlf * 8 + i], q);
  }
  __syncthreads();
#pragma unroll
  for (int i = 0; i < 8; ++i) {
    int r = hlf * 8 + i;
    float mu  = rsum[r] * (1.0f / 512.0f);
    float var = rsq[r] * (1.0f / 512.0f) - mu * mu;
    float rs  = rsqrtf(var + 1e-5f);
#pragma unroll
    for (int t = 0; t < 8; ++t) {
      int col = wave * 128 + t * 16 + lidx;
      float v = (oacc[t][i] - mu) * rs * g1[col] + bet1[col];
      size_t off = (size_t)(m0 + r) * E_ + col;
      midf[off] = v;
      midb[off] = f2bf(v);
    }
  }
}

// ---------------------------------------------------------------------------
// FFN layer 1: h = relu(mid @ W1^T + bf1), bf16 out.  grid (M/32, 2048/256).
// 32-row M-tile, 2 WMMAs per B fragment.
// ---------------------------------------------------------------------------
__global__ void ffn1_gemm(const __bf16* __restrict__ midb,
                          const __bf16* __restrict__ W1b,
                          const float* __restrict__ bias1,
                          __bf16* __restrict__ hb) {
  const int lane = threadIdx.x & 31;
  const int wave = threadIdx.x >> 5;
  const int hlf  = lane >> 4;
  const int lidx = lane & 15;
  const int m0 = blockIdx.x * 32;
  const int n0 = blockIdx.y * 256 + wave * 64;

  f32x8 zero = {};
  f32x8 acc[2][4];
#pragma unroll
  for (int r = 0; r < 2; ++r)
#pragma unroll
    for (int t = 0; t < 4; ++t) acc[r][t] = zero;

#pragma unroll
  for (int k = 0; k < E_; k += 32) {
    bf16x16 a0 = load_frag_a(midb, E_, m0,      k);
    bf16x16 a1 = load_frag_a(midb, E_, m0 + 16, k);
#pragma unroll
    for (int t = 0; t < 4; ++t) {
      bf16x16 bb = load_frag_b(W1b, E_, n0 + t * 16, k);
      acc[0][t] = WMMA_BF16(a0, bb, acc[0][t]);
      acc[1][t] = WMMA_BF16(a1, bb, acc[1][t]);
    }
  }

#pragma unroll
  for (int r = 0; r < 2; ++r)
#pragma unroll
    for (int t = 0; t < 4; ++t) {
      int col = n0 + t * 16 + lidx;
      float bv = bias1[col];
#pragma unroll
      for (int i = 0; i < 8; ++i) {
        float v = fmaxf(acc[r][t][i] + bv, 0.0f);
        hb[(size_t)(m0 + r * 16 + hlf * 8 + i) * F_ + col] = f2bf(v);
      }
    }
}

// ---------------------------------------------------------------------------
// FFN layer 2 + residual + LayerNorm-2 -> f32 output.
// One full 16x512 output row-block per workgroup (wave owns 128 cols).
// ---------------------------------------------------------------------------
__global__ void ffn2_ln(const __bf16* __restrict__ hb,
                        const __bf16* __restrict__ W2b,
                        const float* __restrict__ bias2,
                        const float* __restrict__ midf,
                        const float* __restrict__ g2,
                        const float* __restrict__ bet2,
                        float* __restrict__ out) {
  const int lane = threadIdx.x & 31;
  const int wave = threadIdx.x >> 5;
  const int hlf  = lane >> 4;
  const int lidx = lane & 15;
  const int m0 = blockIdx.x * 16;

  __shared__ float rsum[16], rsq[16];

  f32x8 zero = {};
  f32x8 acc[8];
#pragma unroll
  for (int t = 0; t < 8; ++t) acc[t] = zero;

  for (int k = 0; k < F_; k += 32) {
    bf16x16 a = load_frag_a(hb, F_, m0, k);
#pragma unroll
    for (int t = 0; t < 8; t += 2) {
      bf16x16 b0  = load_frag_b(W2b, F_, wave * 128 + t * 16, k);
      bf16x16 b1v = load_frag_b(W2b, F_, wave * 128 + (t + 1) * 16, k);
      acc[t]     = WMMA_BF16(a, b0, acc[t]);
      acc[t + 1] = WMMA_BF16(a, b1v, acc[t + 1]);
    }
  }

  // bias + residual
#pragma unroll
  for (int t = 0; t < 8; ++t) {
    int col = wave * 128 + t * 16 + lidx;
    float bv = bias2[col];
#pragma unroll
    for (int i = 0; i < 8; ++i) {
      int row = m0 + hlf * 8 + i;
      acc[t][i] += bv + midf[(size_t)row * E_ + col];
    }
  }

  if (threadIdx.x < 16) { rsum[threadIdx.x] = 0.0f; rsq[threadIdx.x] = 0.0f; }
  __syncthreads();
#pragma unroll
  for (int i = 0; i < 8; ++i) {
    float s = 0.0f, q = 0.0f;
#pragma unroll
    for (int t = 0; t < 8; ++t) { float v = acc[t][i]; s += v; q += v * v; }
    atomicAdd(&rsum[hlf * 8 + i], s);
    atomicAdd(&rsq[hlf * 8 + i], q);
  }
  __syncthreads();
#pragma unroll
  for (int i = 0; i < 8; ++i) {
    int r = hlf * 8 + i;
    float mu  = rsum[r] * (1.0f / 512.0f);
    float var = rsq[r] * (1.0f / 512.0f) - mu * mu;
    float rs  = rsqrtf(var + 1e-5f);
#pragma unroll
    for (int t = 0; t < 8; ++t) {
      int col = wave * 128 + t * 16 + lidx;
      out[(size_t)(m0 + r) * E_ + col] = (acc[t][i] - mu) * rs * g2[col] + bet2[col];
    }
  }
}

// ---------------------------------------------------------------------------
// Host launcher
// ---------------------------------------------------------------------------
extern "C" void kernel_launch(void* const* d_in, const int* in_sizes, int n_in,
                              void* d_out, int out_size, void* d_ws, size_t ws_size,
                              hipStream_t stream) {
  (void)in_sizes; (void)n_in; (void)out_size; (void)ws_size;
  const float* x    = (const float*)d_in[0];
  const float* Wv   = (const float*)d_in[1];
  const float* Wk   = (const float*)d_in[2];
  const float* Wq   = (const float*)d_in[3];
  const float* g1   = (const float*)d_in[4];
  const float* b1   = (const float*)d_in[5];
  const float* W1   = (const float*)d_in[6];
  const float* bf1  = (const float*)d_in[7];
  const float* W2   = (const float*)d_in[8];
  const float* bf2  = (const float*)d_in[9];
  const float* g2   = (const float*)d_in[10];
  const float* b2   = (const float*)d_in[11];
  float* out = (float*)d_out;

  char* ws = (char*)d_ws;
  size_t off = 0;
  auto carve = [&](size_t bytes) -> char* {
    char* p = ws + off;
    off += (bytes + 255) & ~(size_t)255;
    return p;
  };
  __bf16* xb   = (__bf16*)carve((size_t)M_ * E_ * 2);
  __bf16* Wqb  = (__bf16*)carve((size_t)E_ * E_ * 2);
  __bf16* Wkb  = (__bf16*)carve((size_t)E_ * E_ * 2);
  __bf16* Wvb  = (__bf16*)carve((size_t)E_ * E_ * 2);
  __bf16* W1b  = (__bf16*)carve((size_t)F_ * E_ * 2);
  __bf16* W2b  = (__bf16*)carve((size_t)E_ * F_ * 2);
  __bf16* qb   = (__bf16*)carve((size_t)M_ * E_ * 2);
  __bf16* kb   = (__bf16*)carve((size_t)M_ * E_ * 2);
  __bf16* vT   = (__bf16*)carve((size_t)B_ * E_ * S_ * 2);
  float*  midf = (float*) carve((size_t)M_ * E_ * 4);
  __bf16* midb = (__bf16*)carve((size_t)M_ * E_ * 2);
  __bf16* hbuf = (__bf16*)carve((size_t)M_ * F_ * 2);

  cvt_f32_bf16<<<(M_ * E_ + 255) / 256, 256, 0, stream>>>(x, xb, M_ * E_);
  cvt_f32_bf16<<<(E_ * E_ + 255) / 256, 256, 0, stream>>>(Wq, Wqb, E_ * E_);
  cvt_f32_bf16<<<(E_ * E_ + 255) / 256, 256, 0, stream>>>(Wk, Wkb, E_ * E_);
  cvt_f32_bf16<<<(E_ * E_ + 255) / 256, 256, 0, stream>>>(Wv, Wvb, E_ * E_);
  cvt_f32_bf16<<<(F_ * E_ + 255) / 256, 256, 0, stream>>>(W1, W1b, F_ * E_);
  cvt_f32_bf16<<<(E_ * F_ + 255) / 256, 256, 0, stream>>>(W2, W2b, E_ * F_);

  qkv_gemm<<<dim3(M_ / 32, E_ / 256, 3), 128, 0, stream>>>(xb, Wqb, Wkb, Wvb, qb, kb, vT);
  attn_flash_ln<<<M_ / 16, 128, 0, stream>>>(qb, kb, vT, x, g1, b1, midf, midb);
  ffn1_gemm<<<dim3(M_ / 32, F_ / 256), 128, 0, stream>>>(midb, W1b, bf1, hbuf);
  ffn2_ln<<<M_ / 16, 128, 0, stream>>>(hbuf, W2b, bf2, midf, g2, b2, out);
}
